// Attention_33809982554295
// MI455X (gfx1250) — matrix operand order repaired and checked by reading, not compiled, not run on hardware
//
#include <hip/hip_runtime.h>
#include <hip/hip_bf16.h>

// ---------------------------------------------------------------------------
// MI455X (gfx1250) attention block, f16 WMMA + async-to-LDS DMA.
//   ~185 GFLOP, ~280 MB traffic -> matrix-op bound. All four GEMMs and both
//   attention matmuls run on V_WMMA_F32_16X16X32_F16 (f32 accumulate).
//   K/V tiles are staged through LDS with GLOBAL_LOAD_ASYNC_TO_LDS_B128,
//   double-buffered and overlapped with WMMA compute (ASYNCcnt pipeline).
// ---------------------------------------------------------------------------

typedef __attribute__((ext_vector_type(16))) _Float16 v16h;
typedef __attribute__((ext_vector_type(8)))  float    v8f;

#define WMMA(a, b, c) \
  __builtin_amdgcn_wmma_f32_16x16x32_f16(false, (a), false, (b), (short)0, (c), false, false)

// ---- WMMA fragment loaders (wave32, this file's contiguous k order, not the ISA table order) ----
// A (16x32 f16): lane = M + 16*khalf; element e -> K = 16*khalf + e.
static __device__ __forceinline__ v16h load_a_frag(const _Float16* A, int lda, int lane) {
  const int m  = lane & 15;
  const int kb = (lane >> 4) << 4;            // 0 or 16
  const _Float16* p = A + (size_t)m * lda + kb;
  v16h r;
#pragma unroll
  for (int e = 0; e < 8; ++e) r[e] = p[e];
#pragma unroll
  for (int e = 0; e < 8; ++e) r[8 + e] = p[8 + e];
  return r;
}

// B (32x16 f16): lane = N + 16*khalf; element e -> K = 16*khalf + e.
static __device__ __forceinline__ v16h load_frag_contig(const _Float16* p) {
  v16h r;
#pragma unroll
  for (int e = 0; e < 16; ++e) r[e] = p[e];
  return r;
}
static __device__ __forceinline__ v16h load_frag_strided(const _Float16* p, int stride) {
  v16h r;
#pragma unroll
  for (int e = 0; e < 16; ++e) r[e] = p[(size_t)e * stride];
  return r;
}

// ---- gfx1250 async DMA: global -> LDS, tracked by ASYNCcnt ----------------
static __device__ __forceinline__ void async_ld_b128(const _Float16* gp, unsigned lds_addr) {
  asm volatile("global_load_async_to_lds_b128 %0, %1, off"
               :: "v"(lds_addr), "v"((unsigned long long)(uintptr_t)gp)
               : "memory");
}

// ---------------------------------------------------------------------------
// Kernel 0: fp32 -> f16 weight conversion
// ---------------------------------------------------------------------------
__global__ void cvt_f32_to_f16_kernel(const float* __restrict__ s,
                                      _Float16* __restrict__ d, int n) {
  int i = blockIdx.x * 256 + threadIdx.x;
  if (i < n) d[i] = (_Float16)s[i];
}

// ---------------------------------------------------------------------------
// Kernel 1: LayerNorm over D=768, one 256-thread block per row, f16 output
// ---------------------------------------------------------------------------
__global__ __launch_bounds__(256) void ln_kernel(const float* __restrict__ y,
                                                 const float* __restrict__ g,
                                                 const float* __restrict__ bt,
                                                 _Float16* __restrict__ xh) {
  const int row = blockIdx.x;
  const float* yr = y + (size_t)row * 768;
  __shared__ float rs[8], rs2[8];
  float v0[3], s = 0.f, s2 = 0.f;
#pragma unroll
  for (int i = 0; i < 3; ++i) {
    float t = yr[threadIdx.x + 256 * i];
    v0[i] = t; s += t; s2 += t * t;
  }
#pragma unroll
  for (int off = 16; off > 0; off >>= 1) {
    s  += __shfl_xor(s,  off, 32);
    s2 += __shfl_xor(s2, off, 32);
  }
  const int wv = threadIdx.x >> 5;
  if ((threadIdx.x & 31) == 0) { rs[wv] = s; rs2[wv] = s2; }
  __syncthreads();
  if (wv == 0) {
    float a  = (threadIdx.x < 8) ? rs[threadIdx.x]  : 0.f;
    float a2 = (threadIdx.x < 8) ? rs2[threadIdx.x] : 0.f;
#pragma unroll
    for (int off = 4; off > 0; off >>= 1) {
      a  += __shfl_xor(a,  off, 32);
      a2 += __shfl_xor(a2, off, 32);
    }
    if (threadIdx.x == 0) { rs[0] = a; rs2[0] = a2; }
  }
  __syncthreads();
  const float mu   = rs[0] * (1.f / 768.f);
  const float var  = rs2[0] * (1.f / 768.f) - mu * mu;
  const float rstd = rsqrtf(var + 1e-6f);
#pragma unroll
  for (int i = 0; i < 3; ++i) {
    const int c = threadIdx.x + 256 * i;
    xh[(size_t)row * 768 + c] = (_Float16)(((v0[i] - mu) * rstd) * g[c] + bt[c]);
  }
}

// ---------------------------------------------------------------------------
// Generic WMMA GEMM: C[M x N] = X[M x K] * W[N x K]^T (+bias)(relu?)(+addin)
// One wave -> 16x64 strip. All fragment loads of a K-step are hoisted ahead
// of the 4 WMMAs so 10 b128 loads are outstanding before the first matmul.
// ---------------------------------------------------------------------------
__global__ __launch_bounds__(128) void gemm_wt_kernel(
    const _Float16* __restrict__ X, int K,
    const _Float16* __restrict__ W,
    const float* __restrict__ bias,
    const float* __restrict__ addin,
    _Float16* __restrict__ outh,
    float* __restrict__ outf,
    int N, int do_relu) {
  const int wid  = blockIdx.x * (blockDim.x >> 5) + (threadIdx.x >> 5);
  const int lane = threadIdx.x & 31;
  const int nstrips = N >> 6;
  const int m0 = (wid / nstrips) * 16;
  const int n0 = (wid % nstrips) * 64;
  const _Float16* Xp = X + (size_t)m0 * K;
  const int n  = lane & 15;
  const int hi = lane >> 4;

  const v8f vz = {0.f, 0.f, 0.f, 0.f, 0.f, 0.f, 0.f, 0.f};
  v8f acc[4] = {vz, vz, vz, vz};

  for (int k0 = 0; k0 < K; k0 += 32) {
    __builtin_prefetch(Xp + k0 + 128, 0, 1);            // global_prefetch_b8
    __builtin_prefetch(W + (size_t)(n0 + n) * K + k0 + 128, 0, 1);
    v16h a = load_a_frag(Xp + k0, K, lane);
    v16h b[4];
#pragma unroll
    for (int j = 0; j < 4; ++j)                          // hoisted B loads
      b[j] = load_frag_contig(W + (size_t)(n0 + j * 16 + n) * K + k0 + (hi << 4));
#pragma unroll
    for (int j = 0; j < 4; ++j)
      acc[j] = WMMA(a, b[j], acc[j]);
  }

#pragma unroll
  for (int j = 0; j < 4; ++j) {
    const int col = n0 + j * 16 + n;
    const float bval = bias ? bias[col] : 0.f;
#pragma unroll
    for (int v = 0; v < 8; ++v) {
      const size_t row = (size_t)(m0 + v + 8 * hi);
      float r = acc[j][v] + bval;
      if (do_relu) r = fmaxf(r, 0.f);
      if (addin) r += addin[row * N + col];
      if (outf) outf[row * N + col] = r;
      if (outh) outh[row * N + col] = (_Float16)r;
    }
  }
}

// ---------------------------------------------------------------------------
// Kernel 3: flash attention. Block = 4 waves sharing one (b,h); each wave
// owns one 16-row Q tile. Wave 0 DMAs the 32x64 K and V chunks into LDS with
// GLOBAL_LOAD_ASYNC_TO_LDS_B128, double-buffered against WMMA compute.
// ---------------------------------------------------------------------------
__global__ __launch_bounds__(128) void attn_kernel(const _Float16* __restrict__ qkv,
                                                   _Float16* __restrict__ ctx_out) {
  __shared__ _Float16 kbuf[2][32 * 64];   // K chunk: rows = key, cols = dh
  __shared__ _Float16 vbuf[2][32 * 64];   // V chunk: rows = key, cols = dh
  __shared__ _Float16 pl[4][16 * 32];     // per-wave P transpose staging

  const int lane = threadIdx.x & 31;
  const int warp = threadIdx.x >> 5;
  const int n  = lane & 15;
  const int hi = lane >> 4;

  const int qblk = blockIdx.x & 31;       // S/64 = 32 blocks per (b,h)
  const int bh   = blockIdx.x >> 5;
  const int h    = bh % 12;
  const int b    = bh / 12;
  const int qt   = qblk * 4 + warp;       // this wave's 16-row Q tile

  const _Float16* Q  = qkv + (size_t)b * 2048 * 2304 + h * 64;
  const _Float16* Kp = Q + 768;
  const _Float16* Vp = Q + 1536;

  // Q tile (16 x 64) as two A fragments, register-resident for the whole loop.
  const v16h qa0 = load_a_frag(Q + (size_t)(qt * 16) * 2304,      2304, lane);
  const v16h qa1 = load_a_frag(Q + (size_t)(qt * 16) * 2304 + 32, 2304, lane);

  // Async DMA of one 32-key K+V chunk (8 KB) -> 16 per-lane b128 issues.
  auto issue_chunk = [&](int tt, int bi) {
#pragma unroll
    for (int i = 0; i < 8; ++i) {
      const int u = i * 32 + lane;             // 256 x 16B units per matrix
      const int row = u >> 3, col = (u & 7) * 8;
      async_ld_b128(Kp + (size_t)(tt + row) * 2304 + col,
                    (unsigned)(uintptr_t)&kbuf[bi][row * 64 + col]);
      async_ld_b128(Vp + (size_t)(tt + row) * 2304 + col,
                    (unsigned)(uintptr_t)&vbuf[bi][row * 64 + col]);
    }
  };

  const v8f vz = {0.f, 0.f, 0.f, 0.f, 0.f, 0.f, 0.f, 0.f};
  v8f acc0 = vz, acc1 = vz, acc2 = vz, acc3 = vz;
  float m_run[8], l_run[8];
#pragma unroll
  for (int v = 0; v < 8; ++v) { m_run[v] = -1e30f; l_run[v] = 0.f; }

  if (warp == 0) issue_chunk(0, 0);       // prologue fill of buffer 0

  int cur = 0;
  for (int t = 0; t < 2048; t += 32) {
    // Pipeline: kick off next chunk's DMA, then wait for current chunk.
    if (warp == 0) {
      if (t + 32 < 2048) {
        issue_chunk(t + 32, cur ^ 1);
        asm volatile("s_wait_asynccnt 16" ::: "memory");  // oldest 16 retired
      } else {
        asm volatile("s_wait_asynccnt 0" ::: "memory");
      }
    }
    __syncthreads();                       // buf[cur] visible to all waves

    const _Float16* kb = kbuf[cur];
    const _Float16* vb = vbuf[cur];

    // --- scores: S = Q K^T * 1/8, tiles (cols t..t+15) and (t+16..t+31)
    v8f s0 = vz, s1 = vz;
    {
      const _Float16* kp0 = kb + n * 64 + (hi << 4);          // key row n
      s0 = WMMA(qa0, load_frag_contig(kp0), s0);
      s0 = WMMA(qa1, load_frag_contig(kp0 + 32), s0);
      const _Float16* kp1 = kb + (16 + n) * 64 + (hi << 4);   // key row 16+n
      s1 = WMMA(qa0, load_frag_contig(kp1), s1);
      s1 = WMMA(qa1, load_frag_contig(kp1 + 32), s1);
    }

    // --- online softmax (C layout: row = v + 8*hi, col = lane&15)
    float alpha[8];
#pragma unroll
    for (int v = 0; v < 8; ++v) {
      float sv0 = s0[v] * 0.125f, sv1 = s1[v] * 0.125f;
      float m = fmaxf(sv0, sv1);
#pragma unroll
      for (int off = 1; off < 16; off <<= 1) m = fmaxf(m, __shfl_xor(m, off, 32));
      const float mn = fmaxf(m_run[v], m);
      const float a  = __expf(m_run[v] - mn);
      const float p0 = __expf(sv0 - mn);
      const float p1 = __expf(sv1 - mn);
      float rsum = p0 + p1;
#pragma unroll
      for (int off = 1; off < 16; off <<= 1) rsum += __shfl_xor(rsum, off, 32);
      l_run[v] = l_run[v] * a + rsum;
      m_run[v] = mn;
      alpha[v] = a;
      pl[warp][(v + 8 * hi) * 32 + n]      = (_Float16)p0;  // P transpose via LDS
      pl[warp][(v + 8 * hi) * 32 + n + 16] = (_Float16)p1;
    }
#pragma unroll
    for (int v = 0; v < 8; ++v) {
      acc0[v] *= alpha[v]; acc1[v] *= alpha[v];
      acc2[v] *= alpha[v]; acc3[v] *= alpha[v];
    }
    asm volatile("s_wait_dscnt 0" ::: "memory");  // DS store -> DS load ordering
    const v16h pa = load_a_frag(pl[warp], 32, lane);

    // --- ctx += P V  (K dim = key index; V columns gathered from LDS)
    const _Float16* vp = vb + (hi << 4) * 64 + n;
    acc0 = WMMA(pa, load_frag_strided(vp,      64), acc0);
    acc1 = WMMA(pa, load_frag_strided(vp + 16, 64), acc1);
    acc2 = WMMA(pa, load_frag_strided(vp + 32, 64), acc2);
    acc3 = WMMA(pa, load_frag_strided(vp + 48, 64), acc3);

    __syncthreads();   // all waves done reading buf[cur] before it's re-filled
    cur ^= 1;
  }

  const size_t orow = (size_t)(b * 2048 + qt * 16);
#pragma unroll
  for (int v = 0; v < 8; ++v) {
    const float inv = 1.f / l_run[v];
    const size_t p = (orow + v + 8 * hi) * 768 + h * 64 + n;
    ctx_out[p]      = (_Float16)(acc0[v] * inv);
    ctx_out[p + 16] = (_Float16)(acc1[v] * inv);
    ctx_out[p + 32] = (_Float16)(acc2[v] * inv);
    ctx_out[p + 48] = (_Float16)(acc3[v] * inv);
  }
}

// ---------------------------------------------------------------------------
// Host-side orchestration
// ---------------------------------------------------------------------------
extern "C" void kernel_launch(void* const* d_in, const int* in_sizes, int n_in,
                              void* d_out, int out_size, void* d_ws, size_t ws_size,
                              hipStream_t stream) {
  (void)in_sizes; (void)n_in; (void)out_size; (void)ws_size;
  const float* y     = (const float*)d_in[0];
  const float* ln_g  = (const float*)d_in[1];
  const float* ln_b  = (const float*)d_in[2];
  const float* qkv_w = (const float*)d_in[3];
  const float* qkv_b = (const float*)d_in[4];
  const float* msa_w = (const float*)d_in[5];
  const float* a1_w  = (const float*)d_in[6];
  const float* a1_b  = (const float*)d_in[7];
  const float* a2_w  = (const float*)d_in[8];
  const float* a2_b  = (const float*)d_in[9];
  float* out = (float*)d_out;

  // Workspace carve-up (all 256B-aligned sizes), total ~208 MB.
  char* w = (char*)d_ws;
  _Float16* xh   = (_Float16*)w; w += (size_t)16384 * 768  * 2;  // LN output
  _Float16* qkvh = (_Float16*)w; w += (size_t)16384 * 2304 * 2;  // QKV output
  _Float16* ctxh = (_Float16*)w; w += (size_t)16384 * 768  * 2;  // attention ctx
  float*    msaf = (float*)w;    w += (size_t)16384 * 768  * 4;  // msa (residual)
  _Float16* msah = (_Float16*)w; w += (size_t)16384 * 768  * 2;  // msa (adapter in)
  _Float16* th   = (_Float16*)w; w += (size_t)16384 * 64   * 2;  // adapter hidden
  _Float16* wqkv = (_Float16*)w; w += (size_t)2304 * 768 * 2;
  _Float16* wmsa = (_Float16*)w; w += (size_t)768  * 768 * 2;
  _Float16* wa1  = (_Float16*)w; w += (size_t)64   * 768 * 2;
  _Float16* wa2  = (_Float16*)w; w += (size_t)768  * 64  * 2;

  // 0) weights -> f16
  cvt_f32_to_f16_kernel<<<6912, 256, 0, stream>>>(qkv_w, wqkv, 2304 * 768);
  cvt_f32_to_f16_kernel<<<2304, 256, 0, stream>>>(msa_w, wmsa, 768 * 768);
  cvt_f32_to_f16_kernel<<<192,  256, 0, stream>>>(a1_w,  wa1,  64 * 768);
  cvt_f32_to_f16_kernel<<<192,  256, 0, stream>>>(a2_w,  wa2,  768 * 64);

  // 1) LayerNorm
  ln_kernel<<<16384, 256, 0, stream>>>(y, ln_g, ln_b, xh);

  // 2) QKV GEMM: 16384x2304 = xh(16384x768) @ qkv_w^T  [36864 waves]
  gemm_wt_kernel<<<9216, 128, 0, stream>>>(xh, 768, wqkv, qkv_b, nullptr,
                                           qkvh, nullptr, 2304, 0);

  // 3) attention: B*H*(S/64) blocks of 4 waves
  attn_kernel<<<3072, 128, 0, stream>>>(qkvh, ctxh);

  // 4) msa = ctx @ msa_w^T (f32 + f16 outputs) [12288 waves]
  gemm_wt_kernel<<<3072, 128, 0, stream>>>(ctxh, 768, wmsa, nullptr, nullptr,
                                           msah, msaf, 768, 0);

  // 5) adapter hidden = relu(msa @ a1_w^T + a1_b) [1024 waves]
  gemm_wt_kernel<<<256, 128, 0, stream>>>(msah, 768, wa1, a1_b, nullptr,
                                          th, nullptr, 64, 1);

  // 6) out = msa + (hidden @ a2_w^T + a2_b) [12288 waves, K=64]
  gemm_wt_kernel<<<3072, 128, 0, stream>>>(th, 64, wa2, a2_b, msaf,
                                           nullptr, out, 768, 0);
}
